// BasicGCN_28776280883362
// MI455X (gfx1250) — matrix operand order repaired and checked by reading, not compiled, not run on hardware
//
#include <hip/hip_runtime.h>
#include <math.h>

#define IN_FEATS 512
#define HIDDEN   256

typedef __attribute__((ext_vector_type(16))) _Float16 v16h;
typedef __attribute__((ext_vector_type(8)))  _Float16 v8h;
typedef __attribute__((ext_vector_type(8)))  float    v8f;

// ---------------- init: zero accumulators, deg=1 (self loop), out=0 ----------------
__global__ void k_init(float* __restrict__ a1, float* __restrict__ deg,
                       float* __restrict__ out, int n_nodes) {
  int t = blockIdx.x * blockDim.x + threadIdx.x;
  if (t < n_nodes * HIDDEN) a1[t] = 0.0f;
  if (t < n_nodes) { deg[t] = 1.0f; out[t] = 0.0f; }
}

// ---------------- in-degree over edges ----------------
__global__ void k_deg(const int* __restrict__ col, float* __restrict__ deg, int E) {
  int e = blockIdx.x * blockDim.x + threadIdx.x;
  if (e < E) atomicAdd(&deg[col[e]], 1.0f);
}

// ---------------- deg -> d^{-1/2} in place ----------------
__global__ void k_dinv(float* __restrict__ deg, int n) {
  int i = blockIdx.x * blockDim.x + threadIdx.x;
  if (i < n) deg[i] = rsqrtf(deg[i]);   // deg >= 1 always
}

// ---------------- GEMM1: h = x @ W1  via WMMA f16->f32, double-buffered LDS ----------------
// block = 256 threads = 8 waves; block tile 128(M) x 64(N); per-wave 16 x 64 strip
#define M_TILE 128
#define N_TILE 64
#define K_TILE 32
#define LDA 40   // halves per A row in LDS (80B stride: bank-conflict free, 16B aligned)
#define LDB 40   // halves per B^T row in LDS

__global__ __launch_bounds__(256) void k_gemm1(const float* __restrict__ X,
                                               const float* __restrict__ W,
                                               float* __restrict__ H, int M) {
  __shared__ _Float16 sA[2][M_TILE * LDA];   // 2 x 10240 B
  __shared__ _Float16 sB[2][N_TILE * LDB];   // 2 x  5120 B

  const int tid  = threadIdx.x;
  const int wave = tid >> 5;
  const int lane = tid & 31;
  const int rowBlk = blockIdx.x * M_TILE;
  const int colBlk = blockIdx.y * N_TILE;
  const bool full = (rowBlk + M_TILE) <= M;   // block-uniform guard

  v8f acc[4];
  const v8f vzero = {0.f,0.f,0.f,0.f,0.f,0.f,0.f,0.f};
#pragma unroll
  for (int t = 0; t < 4; ++t) acc[t] = vzero;

  float4 ra[4];   // staged A tile (this thread's share)
  float  rb[8];   // staged B tile (this thread's share: 4 pairs of consecutive k)

  // global -> registers for K-tile at k0 (uniform fast path when block fully in range)
  auto loadTile = [&](int k0) {
    if (full) {
#pragma unroll
      for (int it = 0; it < 4; ++it) {
        int slot = tid + it * 256;            // 0..1023
        int r  = slot >> 3;                   // 0..127
        int kk = (slot & 7) * 4;              // 0..28
        ra[it] = *(const float4*)(X + (size_t)(rowBlk + r) * IN_FEATS + k0 + kk);
        if (it == 0 && k0 + 2 * K_TILE < IN_FEATS)   // prefetch 2 tiles ahead
          __builtin_prefetch(X + (size_t)(rowBlk + r) * IN_FEATS + k0 + 2 * K_TILE + kk, 0, 1);
      }
    } else {
#pragma unroll
      for (int it = 0; it < 4; ++it) {
        int slot = tid + it * 256;
        int r  = slot >> 3;
        int kk = (slot & 7) * 4;
        int gr = rowBlk + r;
        ra[it] = (gr < M) ? *(const float4*)(X + (size_t)gr * IN_FEATS + k0 + kk)
                          : make_float4(0.f, 0.f, 0.f, 0.f);
      }
    }
    // B: each thread owns 4 (n, k-pair) cells: coalesced over n within two consecutive W rows
#pragma unroll
    for (int it = 0; it < 4; ++it) {
      int p   = tid + it * 256;               // 0..1023 pairs
      int n   = p & 63;                       // coalesced over n
      int kk2 = (p >> 6) * 2;                 // 0,2,...,30
      rb[2 * it + 0] = W[(size_t)(k0 + kk2)     * HIDDEN + colBlk + n];
      rb[2 * it + 1] = W[(size_t)(k0 + kk2 + 1) * HIDDEN + colBlk + n];
    }
  };

  // registers -> LDS buffer `buf` (fp32 -> f16 conversion happens here)
  auto storeTile = [&](int buf) {
#pragma unroll
    for (int it = 0; it < 4; ++it) {
      int slot = tid + it * 256;
      int r  = slot >> 3;
      int kk = (slot & 7) * 4;
      _Float16* dst = &sA[buf][r * LDA + kk];
      dst[0] = (_Float16)ra[it].x; dst[1] = (_Float16)ra[it].y;
      dst[2] = (_Float16)ra[it].z; dst[3] = (_Float16)ra[it].w;
    }
#pragma unroll
    for (int it = 0; it < 4; ++it) {
      int p   = tid + it * 256;
      int n   = p & 63;
      int kk2 = (p >> 6) * 2;
      _Float16* dst = &sB[buf][n * LDB + kk2];     // 4B-aligned pair -> ds_store_b32
      dst[0] = (_Float16)rb[2 * it + 0];
      dst[1] = (_Float16)rb[2 * it + 1];
    }
  };

  loadTile(0);
  storeTile(0);
  __syncthreads();

  int cur = 0;
  for (int k0 = 0; k0 < IN_FEATS; k0 += K_TILE) {
    const bool last = (k0 + K_TILE) >= IN_FEATS;
    if (!last) loadTile(k0 + K_TILE);         // loads in flight during WMMA below

    // A fragment (16x32, 16-bit): lane<16 -> row=lane, k {0..7,16..23}; lane>=16 -> k {8..15,24..31}
    const int ar  = lane & 15;
    const int ako = (lane >> 4) * 8;
    const _Float16* abase = &sA[cur][(wave * 16 + ar) * LDA];
    v8h alo = *(const v8h*)(abase + ako);
    v8h ahi = *(const v8h*)(abase + ako + 16);
    v16h a;
#pragma unroll
    for (int i = 0; i < 8; ++i) { a[i] = alo[i]; a[i + 8] = ahi[i]; }

#pragma unroll
    for (int t = 0; t < 4; ++t) {
      // B fragment (32x16): lane<16 -> col=lane, k {0..7,16..23}; lane>=16 -> col=lane-16, k {8..15,24..31}
      const int bn  = t * 16 + (lane & 15);
      const int bko = (lane >> 4) * 8;
      const _Float16* bbase = &sB[cur][bn * LDB + bko];
      v8h blo = *(const v8h*)(bbase);
      v8h bhi = *(const v8h*)(bbase + 16);
      v16h b;
#pragma unroll
      for (int i = 0; i < 8; ++i) { b[i] = blo[i]; b[i + 8] = bhi[i]; }

      acc[t] = __builtin_amdgcn_wmma_f32_16x16x32_f16(
          /*neg_a=*/false, a, /*neg_b=*/false, b,
          /*c_mod=*/(short)0, acc[t], /*reuse_a=*/false, /*reuse_b=*/false);
    }

    if (!last) {
      storeTile(cur ^ 1);                     // other buffer: no race with readers of `cur`
      __syncthreads();                        // single barrier per K-step
      cur ^= 1;
    }
  }

  // C/D layout: VGPR r, lane l: row = (l>>4)*8 + r, col = l&15 (per 16x16 tile)
  const int mrow = rowBlk + wave * 16 + (lane >> 4) * 8;
  if (full) {
#pragma unroll
    for (int t = 0; t < 4; ++t) {
      int c = colBlk + t * 16 + (lane & 15);
#pragma unroll
      for (int r = 0; r < 8; ++r)
        H[(size_t)(mrow + r) * HIDDEN + c] = acc[t][r];
    }
  } else {
#pragma unroll
    for (int t = 0; t < 4; ++t) {
      int c = colBlk + t * 16 + (lane & 15);
#pragma unroll
      for (int r = 0; r < 8; ++r) {
        int gr = mrow + r;
        if (gr < M) H[(size_t)gr * HIDDEN + c] = acc[t][r];
      }
    }
  }
}

// ---------------- layer-1 edge aggregation: 64 threads/edge, float4 per thread ----------------
__global__ void k_agg1(const int* __restrict__ row, const int* __restrict__ col,
                       const float* __restrict__ dinv, const float* __restrict__ h,
                       float* __restrict__ a1, int E) {
  int t = blockIdx.x * blockDim.x + threadIdx.x;
  int e = t >> 6;
  if (e >= E) return;
  int f = (t & 63) * 4;
  int r = row[e], c = col[e];
  float norm = dinv[r] * dinv[c];
  float4 v = *(const float4*)(h + (size_t)r * HIDDEN + f);
  float* dst = a1 + (size_t)c * HIDDEN + f;
  atomicAdd(dst + 0, v.x * norm);
  atomicAdd(dst + 1, v.y * norm);
  atomicAdd(dst + 2, v.z * norm);
  atomicAdd(dst + 3, v.w * norm);
}

// ---------------- layer-1 self loop + bias + ReLU (non-atomic, stream-ordered) ----------------
__global__ void k_self1(const float* __restrict__ h, const float* __restrict__ dinv,
                        const float* __restrict__ b1, float* __restrict__ a1, int n) {
  int t = blockIdx.x * blockDim.x + threadIdx.x;
  if (t >= n * HIDDEN) return;
  int i = t >> 8;          // HIDDEN == 256
  int f = t & 255;
  float v = a1[t] + h[t] * dinv[i] * dinv[i] + b1[f];
  a1[t] = v > 0.f ? v : 0.f;
}

// ---------------- layer-2 GEMV: h2 = a1 @ W2 (one wave32 per node) ----------------
__global__ void k_gemv2(const float* __restrict__ a1, const float* __restrict__ W2,
                        float* __restrict__ h2, int n) {
  int wid  = (blockIdx.x * blockDim.x + threadIdx.x) >> 5;
  int lane = threadIdx.x & 31;
  if (wid >= n) return;                       // uniform per wave
  const float* rowp = a1 + (size_t)wid * HIDDEN;
  float s = 0.f;
#pragma unroll
  for (int j = 0; j < 8; ++j) { int f = j * 32 + lane; s += rowp[f] * W2[f]; }
#pragma unroll
  for (int m = 16; m > 0; m >>= 1) s += __shfl_xor(s, m, 32);
  if (lane == 0) h2[wid] = s;
}

// ---------------- layer-2 edge aggregation ----------------
__global__ void k_agg2(const int* __restrict__ row, const int* __restrict__ col,
                       const float* __restrict__ dinv, const float* __restrict__ h2,
                       float* __restrict__ out, int E) {
  int e = blockIdx.x * blockDim.x + threadIdx.x;
  if (e < E) {
    int r = row[e], c = col[e];
    atomicAdd(&out[c], h2[r] * dinv[r] * dinv[c]);
  }
}

// ---------------- layer-2 self loop + bias ----------------
__global__ void k_self2(const float* __restrict__ h2, const float* __restrict__ dinv,
                        const float* __restrict__ b2, float* __restrict__ out, int n) {
  int i = blockIdx.x * blockDim.x + threadIdx.x;
  if (i < n) out[i] += h2[i] * dinv[i] * dinv[i] + b2[0];
}

extern "C" void kernel_launch(void* const* d_in, const int* in_sizes, int n_in,
                              void* d_out, int out_size, void* d_ws, size_t ws_size,
                              hipStream_t stream) {
  (void)n_in; (void)out_size; (void)ws_size;
  const float* x  = (const float*)d_in[0];
  const float* W1 = (const float*)d_in[1];
  const float* b1 = (const float*)d_in[2];
  const float* W2 = (const float*)d_in[3];
  const float* b2 = (const float*)d_in[4];
  const int*   ei = (const int*)d_in[5];

  const int N = in_sizes[0] / IN_FEATS;   // 50000
  const int E = in_sizes[5] / 2;          // 800000
  const int* row = ei;
  const int* col = ei + E;

  // workspace carve-out (512B aligned): deg/dinv | h | a1 | h2  (~103 MB)
  char*  ws  = (char*)d_ws;
  size_t off = 0;
  auto carve = [&](size_t bytes) -> float* {
    float* p = (float*)(ws + off);
    off += (bytes + 511) & ~(size_t)511;
    return p;
  };
  float* deg = carve((size_t)N * 4);            // becomes dinv in place
  float* h   = carve((size_t)N * HIDDEN * 4);
  float* a1  = carve((size_t)N * HIDDEN * 4);
  float* h2  = carve((size_t)N * 4);
  float* out = (float*)d_out;

  const int T = 256;
  k_init <<<(N * HIDDEN + T - 1) / T, T, 0, stream>>>(a1, deg, out, N);
  k_deg  <<<(E + T - 1) / T,          T, 0, stream>>>(col, deg, E);
  k_dinv <<<(N + T - 1) / T,          T, 0, stream>>>(deg, N);

  dim3 g1((N + M_TILE - 1) / M_TILE, HIDDEN / N_TILE);
  k_gemm1<<<g1, T, 0, stream>>>(x, W1, h, N);

  k_agg1 <<<((size_t)E * 64 + T - 1) / T, T, 0, stream>>>(row, col, deg, h, a1, E);
  k_self1<<<(N * HIDDEN + T - 1) / T,     T, 0, stream>>>(h, deg, b1, a1, N);

  k_gemv2<<<((size_t)N * 32 + T - 1) / T, T, 0, stream>>>(a1, W2, h2, N);
  k_agg2 <<<(E + T - 1) / T,              T, 0, stream>>>(row, col, deg, h2, out, E);
  k_self2<<<(N + T - 1) / T,              T, 0, stream>>>(h2, deg, b2, out, N);
}
